// SPDAttentionModule_54116587930236
// MI455X (gfx1250) — matrix operand order repaired and checked by reading, not compiled, not run on hardware
//
#include <hip/hip_runtime.h>
#include <cstdint>

// ---------------------------------------------------------------------------
// SPD attention module for MI455X (gfx1250, wave32, WMMA).
// bf16 operands + f32 WMMA accumulation for all GEMMs (workload is
// bandwidth-bound: ~0.55GB traffic floor at 23.3TB/s >> bf16 WMMA time);
// f32 for Cholesky / softmax / blend / log-exp.
// A-operands staged via async global->LDS (ASYNCcnt) into a fragment-swizzled
// layout so every WMMA fragment is read as two contiguous 16B vector loads.
// B-operands are kept/pre-transposed so their fragments are contiguous too.
// ---------------------------------------------------------------------------

constexpr int Bb = 256;   // batch
constexpr int Nn = 256;   // matrix dim
constexpr int Hh = 256;   // hidden
constexpr int NH = 4;     // heads
constexpr int Dd = 64;    // head dim
constexpr float EPSV = 1e-6f;

typedef __attribute__((ext_vector_type(16))) __bf16 v16bf;
typedef __attribute__((ext_vector_type(8)))  __bf16 v8bf;
typedef __attribute__((ext_vector_type(8)))  float  v8f;

#define DEVI static __device__ __forceinline__

// ---- Async global->LDS (CDNA5 GLOBAL_LOAD_ASYNC_TO_LDS_B128) --------------

#if __has_builtin(__builtin_amdgcn_global_load_async_to_lds_b128)
#define HAVE_ASYNC_LDS 1
typedef int v4i_t __attribute__((vector_size(16)));
using as1v4 = __attribute__((address_space(1))) v4i_t;
using as3v4 = __attribute__((address_space(3))) v4i_t;

DEVI void async_cp_b128(const void* gsrc, void* ldst) {
  __builtin_amdgcn_global_load_async_to_lds_b128(
      (as1v4*)(uintptr_t)gsrc, (as3v4*)(uintptr_t)ldst, /*offset=*/0, /*cpol=*/0);
}
DEVI void async_wait0() {
#if __has_builtin(__builtin_amdgcn_s_wait_asynccnt)
  __builtin_amdgcn_s_wait_asynccnt(0);
#else
  asm volatile("s_wait_asynccnt 0x0" ::: "memory");
#endif
}
#else
DEVI void async_wait0() {}
#endif

// ---- Fragment-swizzled A staging ------------------------------------------
// Source: row-major 16x256 bf16 tile (lda=256). Destination LDS layout:
// Afrag[ktile(8)][lane(32)][e(16)] so that lane L of k-step `ktile` reads its
// entire 16-element WMMA A-fragment as 32 contiguous bytes.
// ISA A layout: lane L -> M = L&15, K = k0 + (e/8)*16 + ((L>>4)&1)*8 + (e%8),
// i.e. two contiguous 16B runs per lane in the row-major source.

DEVI void stage_a_swz_issue(const __bf16* __restrict__ gsrc, int lda,
                            __bf16* __restrict__ Afrag, int tid) {
  const int ktile = tid >> 5;         // 0..7  (K step of 32)
  const int lane  = tid & 31;
  const int M     = lane & 15;
  const int hb    = (lane >> 4) & 1;
  const __bf16* src = gsrc + (size_t)M * lda + (ktile << 5) + (hb << 3);
  __bf16*       dst = Afrag + (((ktile << 5) + lane) << 4);
#ifdef HAVE_ASYNC_LDS
  async_cp_b128(src,      dst);       // e = 0..7   (K run k0+hb*8 ..)
  async_cp_b128(src + 16, dst + 8);   // e = 8..15  (K run k0+16+hb*8 ..)
#else
#pragma unroll
  for (int e = 0; e < 8; ++e) { dst[e] = src[e]; dst[e + 8] = src[16 + e]; }
#endif
}

// ---- Vectorized fragment loads --------------------------------------------

DEVI v16bf load2x8(const __bf16* p) {
  v8bf lo = *(const v8bf*)p;
  v8bf hi = *(const v8bf*)(p + 8);
  return __builtin_shufflevector(lo, hi, 0, 1, 2, 3, 4, 5, 6, 7,
                                 8, 9, 10, 11, 12, 13, 14, 15);
}

// A fragment from swizzled LDS (two 16B ds loads).
DEVI v16bf load_a_frag_swz(const __bf16* Afrag, int ktile, int lane) {
  return load2x8(Afrag + (((ktile << 5) + lane) << 4));
}

// B fragment where B[k][n] = Msrc[n][k] (row-major Msrc, contiguous in k):
// lane L -> N = L&15, K = k0 + (e/8)*16 + ((L>>4)&1)*8 + (e%8)  => two 16B runs.
DEVI v16bf load_b_frag_tr(const __bf16* Msrc, int ldm, int k0, int n0, int lane) {
  const int Ncol = lane & 15;
  const int kb   = ((lane >> 4) & 1) << 3;
  const __bf16* p = Msrc + (size_t)(n0 + Ncol) * ldm + k0 + kb;
  v8bf lo = *(const v8bf*)p;
  v8bf hi = *(const v8bf*)(p + 16);
  return __builtin_shufflevector(lo, hi, 0, 1, 2, 3, 4, 5, 6, 7,
                                 8, 9, 10, 11, 12, 13, 14, 15);
}

// ---- WMMA (CDNA5 bf16 16x16x32, f32 accumulate) ---------------------------

DEVI v8f wmma_bf16(v16bf a, v16bf b, v8f c) {
  return __builtin_amdgcn_wmma_f32_16x16x32_bf16(
      /*neg_a=*/false, a, /*neg_b=*/false, b,
      /*c_mod=*/(short)0, c, /*reuse_a=*/false, /*reuse_b=*/false);
}

// ---- K0: weights -> bf16, stored TRANSPOSED (WT[h][n]) --------------------

__global__ __launch_bounds__(256) void wcvt_kernel(
    const float* __restrict__ Wq, const float* __restrict__ Wk,
    __bf16* __restrict__ WqT, __bf16* __restrict__ WkT) {
  const int i = blockIdx.x * 256 + threadIdx.x;   // index into [N][H] row-major
  const int n = i >> 8, h = i & 255;
  WqT[(size_t)h * Nn + n] = (__bf16)Wq[i];
  WkT[(size_t)h * Nn + n] = (__bf16)Wk[i];
}

// ---- K1: symmetrize + LDS Cholesky + logL ---------------------------------
// One workgroup per batch. Packed lower triangle in LDS: 32896 f32 = 128.5KB
// (fits CDNA5's 320KB/WGP LDS). Right-looking rank-1 updates.

__global__ __launch_bounds__(256) void cholesky_logl_kernel(
    const float* __restrict__ spd, float* __restrict__ logL,
    __bf16* __restrict__ logLb) {
  __shared__ float Ls[(Nn * (Nn + 1)) / 2];
  const int b = blockIdx.x, tid = threadIdx.x;
  const float* A = spd + (size_t)b * Nn * Nn;

  for (int i = 0; i < Nn; ++i) {
    int base = (i * (i + 1)) >> 1;
    for (int j = tid; j <= i; j += 256)
      Ls[base + j] = 0.5f * (A[i * Nn + j] + A[j * Nn + i]) + (i == j ? EPSV : 0.0f);
  }
  __syncthreads();

  for (int j = 0; j < Nn; ++j) {
    const int bj = (j * (j + 1)) >> 1;
    if (tid == 0) Ls[bj + j] = sqrtf(fmaxf(Ls[bj + j], 1e-30f));
    __syncthreads();
    const float invd = 1.0f / Ls[bj + j];
    for (int i = j + 1 + tid; i < Nn; i += 256)
      Ls[((i * (i + 1)) >> 1) + j] *= invd;
    __syncthreads();
    for (int i = j + 1 + tid; i < Nn; i += 256) {
      const int bi = (i * (i + 1)) >> 1;
      const float lij = Ls[bi + j];
      for (int k = j + 1; k <= i; ++k)
        Ls[bi + k] = fmaf(-lij, Ls[((k * (k + 1)) >> 1) + j], Ls[bi + k]);
    }
    __syncthreads();
  }

  float*  Lg = logL  + (size_t)b * Nn * Nn;
  __bf16* Lb = logLb + (size_t)b * Nn * Nn;
  for (int i = 0; i < Nn; ++i) {
    int base = (i * (i + 1)) >> 1;
    for (int j = tid; j < Nn; j += 256) {
      float v;
      if (j > i)       v = 0.0f;
      else if (j == i) v = logf(fmaxf(Ls[base + i], EPSV));
      else             v = Ls[base + j];
      Lg[i * Nn + j] = v;
      Lb[i * Nn + j] = (__bf16)v;
    }
  }
}

// ---- K2: q,k = logL @ W + bias (WMMA) -------------------------------------
// Block = (batch, 16-row tile). logL tile async-staged fragment-swizzled.
// Emits q row-major (A for K3), q transposed (B for K6), k row-major (B via
// tr-loads in K3). 8 waves x 4 tasks = 32 output tiles.

__global__ __launch_bounds__(256) void qk_gemm_kernel(
    const __bf16* __restrict__ logLb, const __bf16* __restrict__ WqT,
    const __bf16* __restrict__ WkT, const float* __restrict__ bq,
    const float* __restrict__ bk, __bf16* __restrict__ qout,
    __bf16* __restrict__ qT, __bf16* __restrict__ kout) {
  __shared__ __align__(16) __bf16 Afrag[8 * 32 * 16];   // 8KB
  const int b  = blockIdx.x >> 4;
  const int m0 = (blockIdx.x & 15) << 4;
  const int tid = threadIdx.x, lane = tid & 31, wid = tid >> 5;

  stage_a_swz_issue(logLb + (size_t)b * Nn * Nn + (size_t)m0 * Nn, Nn, Afrag, tid);
  async_wait0();
  __syncthreads();

#pragma unroll
  for (int t = 0; t < 4; ++t) {
    const int task = wid + (t << 3);          // 0..31
    const int isK  = task >> 4;
    const int nt   = task & 15;
    const __bf16* WT   = isK ? WkT : WqT;
    const float*  bias = isK ? bk  : bq;
    v8f c = {};
#pragma unroll
    for (int k0 = 0; k0 < Nn; k0 += 32) {
      v16bf a  = load_a_frag_swz(Afrag, k0 >> 5, lane);
      v16bf bf = load_b_frag_tr(WT, Nn, k0, nt << 4, lane);   // B[k][n]=W[k][n]=WT[n][k]
      c = wmma_bf16(a, bf, c);
    }
    const int col   = (nt << 4) + (lane & 15);
    const int rbase = ((lane >> 4) & 1) << 3;
    const float bv  = bias[col];
    v8bf acc;
#pragma unroll
    for (int r = 0; r < 8; ++r) acc[r] = (__bf16)(c[r] + bv);
    if (isK) {
      __bf16* dst = kout + (size_t)b * Nn * Hh;
#pragma unroll
      for (int r = 0; r < 8; ++r)
        dst[(size_t)(m0 + rbase + r) * Hh + col] = acc[r];
    } else {
      __bf16* dst = qout + (size_t)b * Nn * Hh;
#pragma unroll
      for (int r = 0; r < 8; ++r)
        dst[(size_t)(m0 + rbase + r) * Hh + col] = acc[r];
      // transposed copy: contiguous 16B per-lane vector store
      *(v8bf*)(qT + (size_t)b * Hh * Nn + (size_t)col * Nn + m0 + rbase) = acc;
    }
  }
}

// ---- K3: scores = q @ k^T / temperature (WMMA) ----------------------------

__global__ __launch_bounds__(256) void scores_kernel(
    const __bf16* __restrict__ qb, const __bf16* __restrict__ kbuf,
    const float* __restrict__ temp, float* __restrict__ attnOut) {
  __shared__ __align__(16) __bf16 Qfrag[8 * 32 * 16];   // 8KB
  const int b  = blockIdx.x >> 4;
  const int m0 = (blockIdx.x & 15) << 4;
  const int tid = threadIdx.x, lane = tid & 31, wid = tid >> 5;

  stage_a_swz_issue(qb + (size_t)b * Nn * Hh + (size_t)m0 * Hh, Hh, Qfrag, tid);
  async_wait0();
  __syncthreads();

  const float invT = 1.0f / temp[0];
#pragma unroll
  for (int t = 0; t < 8; ++t) {
    const int task = wid + (t << 3);          // 0..63
    const int h  = task >> 4;
    const int nt = task & 15;
    v8f c = {};
#pragma unroll
    for (int kk = 0; kk < Dd; kk += 32) {
      const int k0 = h * Dd + kk;
      v16bf a  = load_a_frag_swz(Qfrag, k0 >> 5, lane);
      // scores B[k][n] = k[n][h*64+k] -> tr-load on row-major k
      v16bf bf = load_b_frag_tr(kbuf + (size_t)b * Nn * Hh, Hh, k0, nt << 4, lane);
      c = wmma_bf16(a, bf, c);
    }
    float* dst = attnOut + (((size_t)b * NH + h) * Nn + m0) * Nn;
    const int col   = (nt << 4) + (lane & 15);
    const int rbase = ((lane >> 4) & 1) << 3;
#pragma unroll
    for (int r = 0; r < 8; ++r)
      dst[(size_t)(rbase + r) * Nn + col] = c[r] * invT;
  }
}

// ---- K4: row softmax (one wave32 per row, shfl reductions) ----------------

__global__ __launch_bounds__(256) void softmax_kernel(float* __restrict__ attn) {
  const int tid = threadIdx.x, lane = tid & 31, wid = tid >> 5;
  const size_t row = (size_t)blockIdx.x * 8 + wid;   // over B*NH*N rows
  float* p = attn + row * Nn;
  float v[8];
  float mx = -3.4e38f;
#pragma unroll
  for (int t = 0; t < 8; ++t) { v[t] = p[lane + (t << 5)]; mx = fmaxf(mx, v[t]); }
#pragma unroll
  for (int off = 16; off > 0; off >>= 1) mx = fmaxf(mx, __shfl_xor(mx, off, 32));
  float s = 0.0f;
#pragma unroll
  for (int t = 0; t < 8; ++t) { v[t] = __expf(v[t] - mx); s += v[t]; }
#pragma unroll
  for (int off = 16; off > 0; off >>= 1) s += __shfl_xor(s, off, 32);
  const float inv = 1.0f / s;
#pragma unroll
  for (int t = 0; t < 8; ++t) p[lane + (t << 5)] = v[t] * inv;
}

// ---- K5: attn = 0.5*(P + P^T) in place + bf16 copy ------------------------

__global__ __launch_bounds__(256) void sym_attn_kernel(
    float* __restrict__ attn, __bf16* __restrict__ attnb) {
  const int tp = blockIdx.x & 255;
  const int ti = tp >> 4, tj = tp & 15;
  if (ti > tj) return;
  const size_t mat = (size_t)(blockIdx.x >> 8);     // b*NH + h
  float*  P  = attn  + mat * Nn * Nn;
  __bf16* Pb = attnb + mat * Nn * Nn;
  const int r = threadIdx.x >> 4, c = threadIdx.x & 15;
  const int i = ti * 16 + r, j = tj * 16 + c;
  const float x = P[(size_t)i * Nn + j];
  const float y = P[(size_t)j * Nn + i];
  const float avg = 0.5f * (x + y);
  P[(size_t)i * Nn + j] = avg;  P[(size_t)j * Nn + i] = avg;
  Pb[(size_t)i * Nn + j] = (__bf16)avg;  Pb[(size_t)j * Nn + i] = (__bf16)avg;
}

// ---- K6: attended = attn @ q, blended with logL, exp on diag -> E (bf16) --

__global__ __launch_bounds__(256) void attend_blend_kernel(
    const __bf16* __restrict__ attnb, const __bf16* __restrict__ qT,
    const float* __restrict__ logL, const float* __restrict__ blend,
    __bf16* __restrict__ expLb) {
  __shared__ __align__(16) __bf16 Afrag2[2][8 * 32 * 16];   // 16KB
  const int b  = blockIdx.x >> 4;
  const int m0 = (blockIdx.x & 15) << 4;
  const int tid = threadIdx.x, lane = tid & 31, wid = tid >> 5;
  const float w = 1.0f / (1.0f + __expf(-blend[0]));

  for (int hp = 0; hp < 2; ++hp) {
    __syncthreads();
    stage_a_swz_issue(attnb + (((size_t)b * NH + hp * 2 + 0) * Nn + m0) * Nn,
                      Nn, Afrag2[0], tid);
    stage_a_swz_issue(attnb + (((size_t)b * NH + hp * 2 + 1) * Nn + m0) * Nn,
                      Nn, Afrag2[1], tid);
    async_wait0();
    __syncthreads();
    const int h  = hp * 2 + (wid >> 2);
    const int nt = wid & 3;
    v8f c = {};
#pragma unroll
    for (int k0 = 0; k0 < Nn; k0 += 32) {
      v16bf a = load_a_frag_swz(Afrag2[wid >> 2], k0 >> 5, lane);
      // attended B[k][n] = q[k][h*64+n] = qT[h*64+n][k] -> tr-load on qT
      v16bf bf = load_b_frag_tr(qT + (size_t)b * Hh * Nn, Nn, k0,
                                h * Dd + nt * 16, lane);
      c = wmma_bf16(a, bf, c);
    }
    const int col   = h * Dd + nt * 16 + (lane & 15);
    const int rbase = ((lane >> 4) & 1) << 3;
    const float* Lg = logL  + (size_t)b * Nn * Nn;
    __bf16*      E  = expLb + (size_t)b * Nn * Nn;
#pragma unroll
    for (int r = 0; r < 8; ++r) {
      const int row = m0 + rbase + r;
      float u = (1.0f - w) * c[r] + w * Lg[(size_t)row * Nn + col];
      if (row == col) u = __expf(u);
      E[(size_t)row * Nn + col] = (__bf16)u;
    }
  }
}

// ---- K7: spd_out = E @ E^T (WMMA; symmetric by construction) --------------

__global__ __launch_bounds__(256) void spd_out_kernel(
    const __bf16* __restrict__ expLb, float* __restrict__ spdOut) {
  __shared__ __align__(16) __bf16 Efrag[8 * 32 * 16];   // 8KB
  const int b  = blockIdx.x >> 4;
  const int m0 = (blockIdx.x & 15) << 4;
  const int tid = threadIdx.x, lane = tid & 31, wid = tid >> 5;
  const __bf16* E = expLb + (size_t)b * Nn * Nn;

  stage_a_swz_issue(E + (size_t)m0 * Nn, Nn, Efrag, tid);
  async_wait0();
  __syncthreads();

#pragma unroll
  for (int t = 0; t < 2; ++t) {
    const int nt = (wid << 1) + t;
    v8f c = {};
#pragma unroll
    for (int k0 = 0; k0 < Nn; k0 += 32) {
      v16bf a  = load_a_frag_swz(Efrag, k0 >> 5, lane);
      // B[k][n] = E^T[k][n] = E[n][k] -> tr-load on row-major E
      v16bf bf = load_b_frag_tr(E, Nn, k0, nt << 4, lane);
      c = wmma_bf16(a, bf, c);
    }
    float* dst = spdOut + (size_t)b * Nn * Nn;
    const int col   = (nt << 4) + (lane & 15);
    const int rbase = ((lane >> 4) & 1) << 3;
#pragma unroll
    for (int r = 0; r < 8; ++r)
      dst[(size_t)(m0 + rbase + r) * Nn + col] = c[r];
  }
}

// ---- Launch ----------------------------------------------------------------

extern "C" void kernel_launch(void* const* d_in, const int* in_sizes, int n_in,
                              void* d_out, int out_size, void* d_ws, size_t ws_size,
                              hipStream_t stream) {
  const float* spd   = (const float*)d_in[0];
  const float* Wq    = (const float*)d_in[1];
  const float* bq    = (const float*)d_in[2];
  const float* Wk    = (const float*)d_in[3];
  const float* bk    = (const float*)d_in[4];
  const float* temp  = (const float*)d_in[5];
  const float* blend = (const float*)d_in[6];

  float* spdOut  = (float*)d_out;                       // [B,N,N]
  float* attnOut = spdOut + (size_t)Bb * Nn * Nn;       // [B,NH,N,N]

  char* ws = (char*)d_ws;
  size_t off = 0;
  float*  logL  = (float*) (ws + off); off += (size_t)Bb * Nn * Nn * 4;       // 67MB
  __bf16* logLb = (__bf16*)(ws + off); off += (size_t)Bb * Nn * Nn * 2;       // 33.5MB
  __bf16* qbuf  = (__bf16*)(ws + off); off += (size_t)Bb * Nn * Hh * 2;       // 33.5MB
  __bf16* qTb   = (__bf16*)(ws + off); off += (size_t)Bb * Hh * Nn * 2;       // 33.5MB
  __bf16* kbuf  = (__bf16*)(ws + off); off += (size_t)Bb * Nn * Hh * 2;       // 33.5MB
  __bf16* attnb = (__bf16*)(ws + off); off += (size_t)Bb * NH * Nn * Nn * 2;  // 134MB
  __bf16* expLb = (__bf16*)(ws + off); off += (size_t)Bb * Nn * Nn * 2;       // 33.5MB
  __bf16* WqT   = (__bf16*)(ws + off); off += (size_t)Nn * Hh * 2;
  __bf16* WkT   = (__bf16*)(ws + off); off += (size_t)Nn * Hh * 2;

  wcvt_kernel<<<256, 256, 0, stream>>>(Wq, Wk, WqT, WkT);
  cholesky_logl_kernel<<<Bb, 256, 0, stream>>>(spd, logL, logLb);
  qk_gemm_kernel<<<Bb * 16, 256, 0, stream>>>(logLb, WqT, WkT, bq, bk,
                                              qbuf, qTb, kbuf);
  scores_kernel<<<Bb * 16, 256, 0, stream>>>(qbuf, kbuf, temp, attnOut);
  softmax_kernel<<<Bb * NH * Nn / 8, 256, 0, stream>>>(attnOut);
  sym_attn_kernel<<<Bb * NH * 256, 256, 0, stream>>>(attnOut, attnb);
  attend_blend_kernel<<<Bb * 16, 256, 0, stream>>>(attnb, qTb, logL, blend, expLb);
  spd_out_kernel<<<Bb * 16, 256, 0, stream>>>(expLb, spdOut);
}
